// ScalarToDistanceModule_25426206392637
// MI455X (gfx1250) — matrix operand order repaired and checked by reading, not compiled, not run on hardware
//
#include <hip/hip_runtime.h>

// ---------------------------------------------------------------------------
// Edge MLP:  out[e] = W2 . silu( W1 @ concat(s[row[e]], s[col[e]]) + b1 ) + b2
// Batched GEMM M=800000 (edges), K=256, N=128 on the CDNA5 WMMA f16 path
// with fp32 accumulation.  s (25.6MB) is L2-resident, so the kernel is
// compute-bound -> v_wmma_f32_16x16x32_f16 is the right engine.
// ---------------------------------------------------------------------------

typedef _Float16 v16h __attribute__((ext_vector_type(16)));
typedef _Float16 v8h  __attribute__((ext_vector_type(8)));
typedef _Float16 v4h  __attribute__((ext_vector_type(4)));
typedef float    v8f  __attribute__((ext_vector_type(8)));
typedef float    v4f  __attribute__((ext_vector_type(4)));

#define D_S        128
#define TWO_DS     256
#define N_EDGES    800000
#define WAVES      4                     // waves per block (wave32)
#define TILES      2                     // M=16 tiles per wave
#define EDGES_PER_WAVE  (16 * TILES)     // 32
#define EDGES_PER_BLOCK (WAVES * EDGES_PER_WAVE) // 128
#define LDS_STRIDE (TWO_DS + 8)          // 264 halfs = 528B row stride -> 4-bank rotation / row

// LDS layout (in _Float16 units): W1h[128][264] then per-wave A tiles [32][264]
#define W1H_HALFS   (D_S * LDS_STRIDE)
#define ATILE_HALFS (EDGES_PER_WAVE * LDS_STRIDE)
#define SMEM_HALFS  (W1H_HALFS + WAVES * ATILE_HALFS)   // 67584 halfs = 135168 B (<320KB WGP LDS)

__global__ __launch_bounds__(WAVES * 32)
void edge_mlp_wmma(const float* __restrict__ s,
                   const long long* __restrict__ edge_index,   // int64 per reference
                   const float* __restrict__ W1,               // [128, 256] row-major
                   const float* __restrict__ b1,               // [128]
                   const float* __restrict__ W2,               // [1, 128]
                   const float* __restrict__ b2,               // [1]
                   float* __restrict__ out)                    // [E]
{
    extern __shared__ _Float16 smem[];
    _Float16* W1h = smem;

    const int tid  = threadIdx.x;
    const int wave = tid >> 5;
    const int lane = tid & 31;
    const int half = lane >> 4;     // 0: lanes 0-15, 1: lanes 16-31
    const int l16  = lane & 15;

    _Float16* Ah = smem + W1H_HALFS + wave * ATILE_HALFS;

    // ---- Stage W1 fp32 -> f16 into LDS: one row per thread (128 threads, 128 rows)
    {
        const float*  src = W1  + tid * TWO_DS;
        _Float16*     dst = W1h + tid * LDS_STRIDE;
        #pragma unroll
        for (int i = 0; i < TWO_DS; i += 8) {
            v4f f0 = *(const v4f*)(src + i);
            v4f f1 = *(const v4f*)(src + i + 4);
            *(v4h*)(dst + i)     = __builtin_convertvector(f0, v4h);
            *(v4h*)(dst + i + 4) = __builtin_convertvector(f1, v4h);
        }
    }

    // ---- Gather + concat + convert this wave's 32 edge rows into LDS (f16)
    // lanes 0-15 fetch from s[row[e]], lanes 16-31 from s[col[e]]; 8 floats/lane
    const long long base = (long long)blockIdx.x * EDGES_PER_BLOCK
                         + (long long)wave * EDGES_PER_WAVE;
    {
        const long long* rowp = edge_index;             // edge_index[0, :]
        const long long* colp = edge_index + N_EDGES;   // edge_index[1, :]
        for (int m = 0; m < EDGES_PER_WAVE; ++m) {
            const long long e  = base + m;
            const long long nr = half ? colp[e] : rowp[e];
            const float* src = s + nr * D_S + l16 * 8;
            v4f f0 = *(const v4f*)(src);
            v4f f1 = *(const v4f*)(src + 4);
            _Float16* dst = Ah + m * LDS_STRIDE + half * D_S + l16 * 8;
            *(v4h*)(dst)     = __builtin_convertvector(f0, v4h);
            *(v4h*)(dst + 4) = __builtin_convertvector(f1, v4h);
        }
    }
    __syncthreads();

    // ---- Per-lane epilogue constants: lane's N column is j = nb*16 + l16
    float b1v[8], w2v[8];
    #pragma unroll
    for (int nb = 0; nb < 8; ++nb) {
        b1v[nb] = b1[nb * 16 + l16];
        w2v[nb] = W2[nb * 16 + l16];
    }
    const float b2v = b2[0];

    // ---- Main GEMM: 8 K-steps (K=32 each) x 8 N-tiles x TILES M-tiles
    v8f acc[TILES][8];
    #pragma unroll
    for (int t = 0; t < TILES; ++t)
        #pragma unroll
        for (int nb = 0; nb < 8; ++nb)
            acc[t][nb] = (v8f)(0.0f);

    const int hb = half * 16;   // A-fragment K sub-offset (contiguous K, matches B)
    const int kb = half * 16;   // B-fragment K sub-offset (contiguous K, matches A)

    #pragma unroll
    for (int k0 = 0; k0 < 8; ++k0) {
        const int K = k0 * 32;

        // A fragments: lane holds row m=l16; contiguous halfs [hb..hb+15]
        v16h a[TILES];
        #pragma unroll
        for (int t = 0; t < TILES; ++t) {
            const _Float16* ap = Ah + (t * 16 + l16) * LDS_STRIDE + K;
            *((v8h*)&a[t])     = *(const v8h*)(ap + hb);
            *((v8h*)&a[t] + 1) = *(const v8h*)(ap + hb + 8);
        }

        #pragma unroll
        for (int nb = 0; nb < 8; ++nb) {
            // B fragment: lane holds column n=l16 of W1^T => row (nb*16+l16) of W1,
            // contiguous K halfs [kb .. kb+15]
            const _Float16* bp = W1h + (nb * 16 + l16) * LDS_STRIDE + K + kb;
            v16h bfrag;
            *((v8h*)&bfrag)     = *(const v8h*)(bp);
            *((v8h*)&bfrag + 1) = *(const v8h*)(bp + 8);

            #pragma unroll
            for (int t = 0; t < TILES; ++t) {
                acc[t][nb] = __builtin_amdgcn_wmma_f32_16x16x32_f16(
                    /*neg_a=*/false, a[t], /*neg_b=*/false, bfrag,
                    /*c_mod=*/(short)0, acc[t][nb],
                    /*reuse_a=*/false, /*reuse_b=*/false);
            }
        }
    }

    // ---- Epilogue: C/D layout -> lane L, vgpr r holds D[m = r + 8*half][n = l16]
    // out[e] = sum_j silu(h[e][j]) * W2[j] + b2 ; reduce j across 16-lane N groups
    #pragma unroll
    for (int t = 0; t < TILES; ++t) {
        float p[8];
        #pragma unroll
        for (int r = 0; r < 8; ++r) p[r] = 0.0f;

        #pragma unroll
        for (int nb = 0; nb < 8; ++nb) {
            #pragma unroll
            for (int r = 0; r < 8; ++r) {
                const float x  = acc[t][nb][r] + b1v[nb];
                const float sg = 1.0f / (1.0f + __expf(-x));
                p[r] += (x * sg) * w2v[nb];
            }
        }

        // butterfly sum across the 16 lanes sharing the same edge set
        #pragma unroll
        for (int r = 0; r < 8; ++r) {
            #pragma unroll
            for (int mask = 1; mask <= 8; mask <<= 1)
                p[r] += __shfl_xor(p[r], mask, 32);
        }

        if (l16 == 0) {
            const long long eb = base + t * 16 + half * 8;
            #pragma unroll
            for (int r = 0; r < 8; ++r)
                out[eb + r] = p[r] + b2v;
        }
    }
}

extern "C" void kernel_launch(void* const* d_in, const int* in_sizes, int n_in,
                              void* d_out, int out_size, void* d_ws, size_t ws_size,
                              hipStream_t stream) {
    const float*     s  = (const float*)d_in[0];
    const long long* ei = (const long long*)d_in[1];   // int64 edge_index [2, E]
    const float*     W1 = (const float*)d_in[2];
    const float*     b1 = (const float*)d_in[3];
    const float*     W2 = (const float*)d_in[4];
    const float*     b2 = (const float*)d_in[5];
    float* out = (float*)d_out;

    const int shmem = SMEM_HALFS * (int)sizeof(_Float16);   // 135168 B (>64KB: CDNA5 WGP LDS)
    (void)hipFuncSetAttribute(reinterpret_cast<const void*>(edge_mlp_wmma),
                              hipFuncAttributeMaxDynamicSharedMemorySize, shmem);

    const int blocks = (N_EDGES + EDGES_PER_BLOCK - 1) / EDGES_PER_BLOCK;  // 6250
    edge_mlp_wmma<<<blocks, WAVES * 32, shmem, stream>>>(s, ei, W1, b1, W2, b2, out);
}
